// SequentialRNN_53738630808120
// MI455X (gfx1250) — matrix operand order repaired and checked by reading, not compiled, not run on hardware
//
#include <hip/hip_runtime.h>
#include <math.h>

#define N_SAMP 256
#define E_DIM  2048
#define H_DIM  2048
#define G_DIM  8192   // 4*H
#define T_STEPS 768   // 3*N
#define K_E    2048

typedef __attribute__((ext_vector_type(16))) __bf16 v16bf;
typedef __attribute__((ext_vector_type(8)))  __bf16 v8bf;
typedef __attribute__((ext_vector_type(8)))  float  v8f;

__device__ __forceinline__ unsigned short f2bf_bits(float f) {
  unsigned u = __builtin_bit_cast(unsigned, f);
  u += 0x7fffu + ((u >> 16) & 1u);            // round-to-nearest-even
  return (unsigned short)(u >> 16);
}

// ---------------------------------------------------------------------------
// Init: zero barrier counter + h double-buffer (h0 = 0). Re-run every launch
// so graph replays are deterministic.
// ---------------------------------------------------------------------------
__global__ void init_kernel(float* hbuf, unsigned* cnt) {
  if (threadIdx.x == 0) *cnt = 0u;
  for (int k = threadIdx.x; k < 2 * H_DIM; k += 256) hbuf[k] = 0.0f;
}

// ---------------------------------------------------------------------------
// Embedding gather, emitted directly as packed bf16 pairs:
// X_bf[t, k..k+1], t = 3n+s  (s=0: ctx, s=1: f1, s=2: f2). One u32 per thread.
// ---------------------------------------------------------------------------
__global__ void embed_bf16_kernel(const int* __restrict__ ctx, const int* __restrict__ f1,
                                  const int* __restrict__ f2,
                                  const float* __restrict__ ctx_table,
                                  const float* __restrict__ face_table,
                                  unsigned* __restrict__ X_bf) {
  int pair = blockIdx.x * 256 + threadIdx.x;      // 0 .. T*E/2-1 exactly
  int t  = pair >> 10;                            // / (E/2)
  int k  = (pair & (E_DIM / 2 - 1)) * 2;
  int n = t / 3, s = t - 3 * n;
  const float* src;
  if (s == 0)      src = ctx_table  + (size_t)ctx[n] * E_DIM;   // CTX_SCALE = 1.0
  else if (s == 1) src = face_table + (size_t)f1[n] * E_DIM;
  else             src = face_table + (size_t)f2[n] * E_DIM;
  float lo = src[k], hi = src[k + 1];
  X_bf[pair] = ((unsigned)f2bf_bits(hi) << 16) | (unsigned)f2bf_bits(lo);
}

// ---------------------------------------------------------------------------
// Generic fp32 -> packed bf16-pair conversion (used for W_ih and W_hh).
// ---------------------------------------------------------------------------
__global__ void pack_bf16_kernel(const float* __restrict__ src, unsigned* __restrict__ dst) {
  int idx = blockIdx.x * 256 + threadIdx.x;       // grid sized exactly
  float lo = src[2 * (size_t)idx];
  float hi = src[2 * (size_t)idx + 1];
  dst[idx] = ((unsigned)f2bf_bits(hi) << 16) | (unsigned)f2bf_bits(lo);
}

// ---------------------------------------------------------------------------
// Pre-gate GEMM: Gpre(768 x 8192) = X(768x2048) @ W_ih^T + (b_ih + b_hh)
// bf16 inputs, fp32 accumulate. Each wave owns a 16x64 strip (4 N-tiles,
// A reused 4x). The 8 waves of a block share the SAME 64-column B strip
// (identical B addresses -> WGP$ reuse of the 32 MB operand) and span 8
// consecutive M tiles. Fragments for step kt+1 are prefetched into registers
// before the WMMAs of step kt, so the matrix ops never wait on loadcnt==0.
// Fragment layouts per CDNA5 ISA 7.12.2.
// ---------------------------------------------------------------------------
struct Frags { v16bf a, b0, b1, b2, b3; };

__device__ __forceinline__ v16bf load_a_frag(const __bf16* __restrict__ arow, int kt, int half) {
  const int abase = kt * 32 + half * 8;           // A: K {base..+7, base+16..+23}
  v8bf lo = *(const v8bf*)(arow + abase);
  v8bf hi = *(const v8bf*)(arow + abase + 16);
  return __builtin_shufflevector(lo, hi, 0,1,2,3,4,5,6,7,8,9,10,11,12,13,14,15);
}
__device__ __forceinline__ v16bf load_b_frag(const __bf16* __restrict__ brow, int kt, int half) {
  const int bbase = kt * 32 + half * 8;           // B: K {base..+7, base+16..+23}
  v8bf lo = *(const v8bf*)(brow + bbase);
  v8bf hi = *(const v8bf*)(brow + bbase + 16);
  return __builtin_shufflevector(lo, hi, 0,1,2,3,4,5,6,7,8,9,10,11,12,13,14,15);
}

__global__ void __launch_bounds__(256)
pregate_wmma_kernel(const __bf16* __restrict__ Xb, const __bf16* __restrict__ Wb,
                    const float* __restrict__ b_ih, const float* __restrict__ b_hh,
                    float* __restrict__ Gpre) {
  const int wave = threadIdx.x >> 5;
  const int lane = threadIdx.x & 31;
  const int tile_n4 = blockIdx.x & 127;           // B strip shared by all 8 waves
  const int tile_m  = (blockIdx.x >> 7) * 8 + wave;   // 0..47 (6 M-blocks)
  const int half = lane >> 4;                     // 0: lanes 0-15, 1: lanes 16-31
  const int l16  = lane & 15;

  const __bf16* __restrict__ arow  = Xb + (size_t)(tile_m * 16 + l16) * K_E;
  const __bf16* __restrict__ brow0 = Wb + (size_t)(tile_n4 * 64 + 0 * 16 + l16) * K_E;
  const __bf16* __restrict__ brow1 = Wb + (size_t)(tile_n4 * 64 + 1 * 16 + l16) * K_E;
  const __bf16* __restrict__ brow2 = Wb + (size_t)(tile_n4 * 64 + 2 * 16 + l16) * K_E;
  const __bf16* __restrict__ brow3 = Wb + (size_t)(tile_n4 * 64 + 3 * 16 + l16) * K_E;

  v8f acc0 = {}, acc1 = {}, acc2 = {}, acc3 = {};

  Frags cur;
  cur.a  = load_a_frag(arow,  0, half);
  cur.b0 = load_b_frag(brow0, 0, half);
  cur.b1 = load_b_frag(brow1, 0, half);
  cur.b2 = load_b_frag(brow2, 0, half);
  cur.b3 = load_b_frag(brow3, 0, half);

  for (int kt = 0; kt < K_E / 32; ++kt) {
    const int kn = (kt + 1) & (K_E / 32 - 1);     // wrap: last prefetch is harmless
    Frags nxt;
    nxt.a  = load_a_frag(arow,  kn, half);
    nxt.b0 = load_b_frag(brow0, kn, half);
    nxt.b1 = load_b_frag(brow1, kn, half);
    nxt.b2 = load_b_frag(brow2, kn, half);
    nxt.b3 = load_b_frag(brow3, kn, half);

    acc0 = __builtin_amdgcn_wmma_f32_16x16x32_bf16(false, cur.a, false, cur.b0, (short)0, acc0, false, false);
    acc1 = __builtin_amdgcn_wmma_f32_16x16x32_bf16(false, cur.a, false, cur.b1, (short)0, acc1, false, false);
    acc2 = __builtin_amdgcn_wmma_f32_16x16x32_bf16(false, cur.a, false, cur.b2, (short)0, acc2, false, false);
    acc3 = __builtin_amdgcn_wmma_f32_16x16x32_bf16(false, cur.a, false, cur.b3, (short)0, acc3, false, false);
    cur = nxt;
  }

#pragma unroll
  for (int q = 0; q < 4; ++q) {
    const v8f acc = (q == 0) ? acc0 : (q == 1) ? acc1 : (q == 2) ? acc2 : acc3;
    const int jcol = tile_n4 * 64 + q * 16 + l16;
    const float bias = b_ih[jcol] + b_hh[jcol];
#pragma unroll
    for (int r = 0; r < 8; ++r) {                 // C/D: VGPR r -> M = half*8 + r
      int row = tile_m * 16 + half * 8 + r;
      Gpre[(size_t)row * G_DIM + jcol] = acc[r] + bias;
    }
  }
}

// ---------------------------------------------------------------------------
// Persistent recurrent kernel. 256 blocks x 256 threads.
// Block b owns hidden units jh = 8b..8b+7 => gate rows g*2048 + jh (32 rows).
// c lives in registers (threads 0..7); h is double-buffered in global memory;
// one device-wide barrier per step (monotonic atomic counter).
// W_hh is bf16 -> 32 MB/step streamed from the 192 MB L2.
// ---------------------------------------------------------------------------
__global__ void __launch_bounds__(256)
lstm_persistent_kernel(const unsigned* __restrict__ whh_bf,
                       const float* __restrict__ Gpre,
                       float* __restrict__ hbuf,     // 2*H double buffer
                       float* __restrict__ cF,
                       float* __restrict__ Hsamp,    // N x H (h after each sample)
                       unsigned* __restrict__ cnt) {
  __shared__ float lds_h[H_DIM];
  __shared__ float lds_dot[32];
  const int tid  = threadIdx.x;
  const int wave = tid >> 5;
  const int lane = tid & 31;
  const unsigned nb = gridDim.x;

  float creg = 0.0f;                            // c0 = 0 (threads 0..7 use it)

  for (int t = 0; t < T_STEPS; ++t) {
    const float* __restrict__ hread = hbuf + (t & 1) * H_DIM;
    float* __restrict__ hwrite      = hbuf + ((t + 1) & 1) * H_DIM;

    // stage h_t into LDS (8 KB)
    for (int k = tid; k < H_DIM; k += 256) lds_h[k] = hread[k];
    __syncthreads();

    // wave handles rows wave*4 .. wave*4+3 (row_local = gate*8 + rl)
#pragma unroll
    for (int r = 0; r < 4; ++r) {
      const int row_local = wave * 4 + r;
      const int gate = row_local >> 3;
      const int rl   = row_local & 7;
      const int row  = gate * H_DIM + blockIdx.x * 8 + rl;
      const unsigned* __restrict__ wr = whh_bf + (size_t)row * (H_DIM / 2);
      float s = 0.0f;
      for (int k2 = lane; k2 < H_DIM / 2; k2 += 32) {   // coalesced 128B/wave
        unsigned wb = wr[k2];
        float w0 = __builtin_bit_cast(float, wb << 16);          // bf16 -> f32 exact
        float w1 = __builtin_bit_cast(float, wb & 0xffff0000u);
        s += w0 * lds_h[2 * k2] + w1 * lds_h[2 * k2 + 1];
      }
#pragma unroll
      for (int off = 16; off > 0; off >>= 1) s += __shfl_xor(s, off, 32);
      if (lane == 0) lds_dot[row_local] = s;
    }
    __syncthreads();

    if (tid < 8) {
      const int jh = blockIdx.x * 8 + tid;
      const float* __restrict__ gp = Gpre + (size_t)t * G_DIM;
      float gi = gp[0 * H_DIM + jh] + lds_dot[0 * 8 + tid];
      float gf = gp[1 * H_DIM + jh] + lds_dot[1 * 8 + tid];
      float gg = gp[2 * H_DIM + jh] + lds_dot[2 * 8 + tid];
      float go = gp[3 * H_DIM + jh] + lds_dot[3 * 8 + tid];
      float iv = 1.0f / (1.0f + __expf(-gi));
      float fv = 1.0f / (1.0f + __expf(-gf));
      float gv = tanhf(gg);
      float ov = 1.0f / (1.0f + __expf(-go));
      creg = fv * creg + iv * gv;
      float hv = ov * tanhf(creg);
      hwrite[jh] = hv;
      if (t % 3 == 2) Hsamp[(size_t)(t / 3) * H_DIM + jh] = hv;
      if (t == T_STEPS - 1) cF[jh] = creg;
    }

    // device-wide barrier (release/acquire around monotonic counter)
    __threadfence();
    __syncthreads();
    if (tid == 0) {
      atomicAdd(cnt, 1u);
      const unsigned target = nb * (unsigned)(t + 1);
      while (atomicAdd(cnt, 0u) < target) __builtin_amdgcn_s_sleep(8);
    }
    __syncthreads();
    __threadfence();
  }
}

// ---------------------------------------------------------------------------
// Output head + pack d_out = [outputs(256x2) | hF(2048) | cF(2048)]
// Final h lives in hbuf buffer 0 (step 767 writes buffer (768&1)=0).
// ---------------------------------------------------------------------------
__global__ void finalize_kernel(const float* __restrict__ Hsamp,
                                const float* __restrict__ hbuf0,
                                const float* __restrict__ cF,
                                const float* __restrict__ Wout,
                                const float* __restrict__ bout,
                                float* __restrict__ out) {
  const int tid = threadIdx.x;
  if (blockIdx.x == 0) {
    const int n = tid;                           // 256 threads = 256 samples
    const float* hv = Hsamp + (size_t)n * H_DIM;
    float s0 = 0.0f, s1 = 0.0f;
    for (int k = 0; k < H_DIM; ++k) {
      float h = hv[k];
      s0 += Wout[k] * h;
      s1 += Wout[H_DIM + k] * h;
    }
    out[n * 2 + 0] = s0 + bout[0];
    out[n * 2 + 1] = s1 + bout[1];
  } else {
    const int idx = (blockIdx.x - 1) * 256 + tid;  // 0..2047
    out[2 * N_SAMP + idx]         = hbuf0[idx];    // hF
    out[2 * N_SAMP + H_DIM + idx] = cF[idx];       // cF
  }
}

// ---------------------------------------------------------------------------
extern "C" void kernel_launch(void* const* d_in, const int* in_sizes, int n_in,
                              void* d_out, int out_size, void* d_ws, size_t ws_size,
                              hipStream_t stream) {
  const int*   ctx        = (const int*)  d_in[0];
  const int*   f1         = (const int*)  d_in[1];
  const int*   f2         = (const int*)  d_in[2];
  const float* ctx_table  = (const float*)d_in[3];
  const float* face_table = (const float*)d_in[4];
  const float* W_ih       = (const float*)d_in[5];
  const float* W_hh       = (const float*)d_in[6];
  const float* b_ih       = (const float*)d_in[7];
  const float* b_hh       = (const float*)d_in[8];
  const float* W_out      = (const float*)d_in[9];
  const float* b_out      = (const float*)d_in[10];
  float* out = (float*)d_out;
  (void)in_sizes; (void)n_in; (void)out_size; (void)ws_size;

  char* base = (char*)d_ws;
  size_t off = 0;
  auto take = [&](size_t bytes) -> char* {
    char* p = base + off;
    off += (bytes + 255) & ~(size_t)255;
    return p;
  };
  unsigned* cnt   = (unsigned*)take(sizeof(unsigned));
  unsigned* Xbf   = (unsigned*)take((size_t)T_STEPS * (E_DIM / 2) * sizeof(unsigned));  //  3 MB
  unsigned* WihBf = (unsigned*)take((size_t)G_DIM * (K_E / 2) * sizeof(unsigned));      // 32 MB
  unsigned* WhhBf = (unsigned*)take((size_t)G_DIM * (H_DIM / 2) * sizeof(unsigned));    // 32 MB
  float*    Gpre  = (float*)   take((size_t)T_STEPS * G_DIM * sizeof(float));           // 25 MB
  float*    hbuf  = (float*)   take((size_t)2 * H_DIM * sizeof(float));
  float*    cFbuf = (float*)   take((size_t)H_DIM * sizeof(float));
  float*    Hsamp = (float*)   take((size_t)N_SAMP * H_DIM * sizeof(float));            //  2 MB

  init_kernel<<<1, 256, 0, stream>>>(hbuf, cnt);
  embed_bf16_kernel<<<(T_STEPS * E_DIM / 2) / 256, 256, 0, stream>>>(
      ctx, f1, f2, ctx_table, face_table, Xbf);
  pack_bf16_kernel<<<(G_DIM * (K_E / 2)) / 256, 256, 0, stream>>>(W_ih, WihBf);
  pack_bf16_kernel<<<(G_DIM * (H_DIM / 2)) / 256, 256, 0, stream>>>(W_hh, WhhBf);
  pregate_wmma_kernel<<<(48 * 128) / 8, 256, 0, stream>>>(
      (const __bf16*)Xbf, (const __bf16*)WihBf, b_ih, b_hh, Gpre);
  lstm_persistent_kernel<<<256, 256, 0, stream>>>(WhhBf, Gpre, hbuf, cFbuf, Hsamp, cnt);
  finalize_kernel<<<9, 256, 0, stream>>>(Hsamp, hbuf /*buffer 0*/, cFbuf, W_out, b_out, out);
}
